// EMAVectorQuantizer_71047349010675
// MI455X (gfx1250) — matrix operand order repaired and checked by reading, not compiled, not run on hardware
//
#include <hip/hip_runtime.h>
#include <hip/hip_bf16.h>

// ---------------------------------------------------------------------------
// EMA Vector Quantizer for MI455X (gfx1250, wave32, WMMA + double-buffered
// async global->LDS staging + double-buffered B operand register tiles).
//
// Sizes (from reference):
//   z:          [8,4096,256]  -> N = 32768 rows, D = 256
//   embeddings: [8192,256]    -> K = 8192
//
// Output layout (floats, concatenated in return order):
//   [0]          z_q_st            8*4096*256 = 8388608
//   [8388608]    indices (as f32)  32768
//   [8421376]    commitment_loss   1
//   [8421377]    new_embeddings    2097152
//   [10518529]   new_ema_cluster   8192
//   [10526721]   new_ema_embed_sum 2097152
// ---------------------------------------------------------------------------

#define N_ROWS 32768
#define DIM    256
#define K_CODE 8192
#define NCHUNK (K_CODE / 64)        // 128 chunks of 64 codebook rows
#define CHUNK_BYTES (64 * DIM * 2)  // 32 KB of bf16 per chunk

typedef __attribute__((ext_vector_type(16))) __bf16 bf16x16;
typedef __attribute__((ext_vector_type(8)))  float  f32x8;

// ---------------------------------------------------------------------------
// Kernel 1: prep — convert embeddings to bf16 (ws), compute e_sq, zero the
// scatter accumulators (which live in the output regions) and the loss.
// grid = K_CODE blocks of 256 threads (one block per codebook row).
// ---------------------------------------------------------------------------
__global__ void vq_prep(const float* __restrict__ E,
                        __bf16* __restrict__ Eb,
                        float* __restrict__ e_sq,
                        float* __restrict__ out_cs,   // scatter: counts
                        float* __restrict__ out_es,   // scatter: embed sums
                        float* __restrict__ loss_ws)
{
    const int k = blockIdx.x;
    const int d = threadIdx.x;
    const int i = k * DIM + d;
    float v = E[i];
    Eb[i] = (__bf16)v;
    out_es[i] = 0.0f;                 // zero scatter accumulator

    __shared__ float red[256];
    red[d] = v * v;
    __syncthreads();
    #pragma unroll
    for (int s = 128; s > 0; s >>= 1) {
        if (d < s) red[d] += red[d + s];
        __syncthreads();
    }
    if (d == 0) {
        e_sq[k] = red[0];
        out_cs[k] = 0.0f;
        if (k == 0) loss_ws[0] = 0.0f;
    }
}

// ---------------------------------------------------------------------------
// Kernel 2: fused distance-matmul + row argmin using v_wmma_f32_16x16x32_bf16.
//
// Each wave owns a 16-row tile of z; the full 16x256 bf16 A matrix is held in
// 64 VGPRs (8 x bf16x16), ISA A layout. The block (8 waves) streams the
// codebook through a 2x32KB double-buffered LDS window filled with
// GLOBAL_LOAD_ASYNC_TO_LDS_B128 (ASYNCcnt): the copy of chunk i+1 overlaps
// the 32 WMMAs of chunk i. Inside a chunk, the 4 B tiles of k-step c+1 are
// loaded into a second register buffer before the 4 WMMAs of k-step c issue,
// so the DS waits become partial and LDS latency hides under XDL execution.
//
// C layout: lane L holds column N=L&15; VGPR r holds row M=r+8*hi.
// Score = e_sq[n] - 2*dot  (z_sq is row-constant, irrelevant for argmin).
// ---------------------------------------------------------------------------
__global__ void __launch_bounds__(256, 1)
vq_argmin(const float*  __restrict__ z,
          const __bf16* __restrict__ Eb,
          const float*  __restrict__ e_sq,
          int*          __restrict__ idx_ws,
          float*        __restrict__ idx_out_f)
{
    const int lane = threadIdx.x & 31;
    const int wave = threadIdx.x >> 5;
    const int lo   = lane & 15;
    const int hi   = lane >> 4;
    const int rowBase = blockIdx.x * 128 + wave * 16;

    __shared__ __align__(16) __bf16 ldsE[2 * 64 * DIM];   // 2 x 32 KB

    // ---- load A (16 rows x 256) into registers as bf16, ISA layout ----
    bf16x16 a[8];
    const float* zr = z + (size_t)(rowBase + lo) * DIM;
    #pragma unroll
    for (int c = 0; c < 8; ++c) {
        const int k0 = c * 32 + hi * 8;
        float4 f0 = *(const float4*)(zr + k0);
        float4 f1 = *(const float4*)(zr + k0 + 4);
        float4 g0 = *(const float4*)(zr + k0 + 16);
        float4 g1 = *(const float4*)(zr + k0 + 20);
        a[c][0]  = (__bf16)f0.x;  a[c][1]  = (__bf16)f0.y;
        a[c][2]  = (__bf16)f0.z;  a[c][3]  = (__bf16)f0.w;
        a[c][4]  = (__bf16)f1.x;  a[c][5]  = (__bf16)f1.y;
        a[c][6]  = (__bf16)f1.z;  a[c][7]  = (__bf16)f1.w;
        a[c][8]  = (__bf16)g0.x;  a[c][9]  = (__bf16)g0.y;
        a[c][10] = (__bf16)g0.z;  a[c][11] = (__bf16)g0.w;
        a[c][12] = (__bf16)g1.x;  a[c][13] = (__bf16)g1.y;
        a[c][14] = (__bf16)g1.z;  a[c][15] = (__bf16)g1.w;
    }

    float minv[8];
    int   mini[8];
    #pragma unroll
    for (int r = 0; r < 8; ++r) { minv[r] = 3.4e38f; mini[r] = 0; }

    // LDS byte offset of the staging buffers (flat addr low 32 bits == LDS addr)
    const unsigned ldsBase = (unsigned)(size_t)(void*)ldsE;

    // Issue the 8 b128 async transfers this thread owns for one chunk.
    auto issue_chunk = [&](int chunk, int bufsel) {
        unsigned long long gbase = (unsigned long long)(size_t)Eb
                                 + (unsigned long long)chunk * CHUNK_BYTES;
        unsigned base = ldsBase + (unsigned)bufsel * (unsigned)CHUNK_BYTES;
        #pragma unroll
        for (int it = 0; it < 8; ++it) {
            unsigned voff = (threadIdx.x + it * 256) * 16u;   // byte offset
            unsigned dlds = base + voff;
            asm volatile("global_load_async_to_lds_b128 %0, %1, %2"
                         :: "v"(dlds), "v"(voff), "s"(gbase) : "memory");
        }
    };

    issue_chunk(0, 0);

    for (int i = 0; i < NCHUNK; ++i) {
        if (i + 1 < NCHUNK) {
            issue_chunk(i + 1, (i + 1) & 1);
            // older chunk's 8 transfers complete in order -> wait to <= 8
            asm volatile("s_wait_asynccnt 0x8" ::: "memory");
        } else {
            asm volatile("s_wait_asynccnt 0x0" ::: "memory");
        }
        __syncthreads();   // chunk i resident for all waves

        const __bf16* buf = ldsE + (i & 1) * (64 * DIM);
        const int nbase = i * 64;

        // hoist e_sq loads so they overlap the WMMA chain
        float es[4];
        #pragma unroll
        for (int t = 0; t < 4; ++t) es[t] = e_sq[nbase + t * 16 + lo];

        // B tile loader: (c,t) -> two 8-bf16 runs (k=8hi.., 16+8hi..), ISA B layout
        auto loadB = [&](int c, int t) -> bf16x16 {
            typedef __attribute__((ext_vector_type(8))) __bf16 bf16x8;
            const __bf16* brow = buf + (t * 16 + lo) * DIM + c * 32 + hi * 8;
            bf16x8 p0 = *(const bf16x8*)(brow);
            bf16x8 p1 = *(const bf16x8*)(brow + 16);
            return __builtin_shufflevector(p0, p1, 0,1,2,3,4,5,6,7,8,9,10,11,12,13,14,15);
        };

        // k-step double buffer: load all 4 B tiles of step c+1 before the
        // 4 WMMAs of step c issue -> partial DS waits, latency under XDL.
        f32x8 acc[4] = {{}, {}, {}, {}};
        bf16x16 b0[4], b1[4];
        #pragma unroll
        for (int t = 0; t < 4; ++t) b0[t] = loadB(0, t);
        #pragma unroll
        for (int c = 0; c < 8; ++c) {
            bf16x16* bc = (c & 1) ? b1 : b0;
            bf16x16* bn = (c & 1) ? b0 : b1;
            if (c < 7) {
                #pragma unroll
                for (int t = 0; t < 4; ++t) bn[t] = loadB(c + 1, t);
            }
            #pragma unroll
            for (int t = 0; t < 4; ++t) {
                acc[t] = __builtin_amdgcn_wmma_f32_16x16x32_bf16(
                            false, a[c], false, bc[t], (short)0, acc[t],
                            false, false);
            }
        }

        #pragma unroll
        for (int t = 0; t < 4; ++t) {
            const int n = nbase + t * 16 + lo;
            #pragma unroll
            for (int r = 0; r < 8; ++r) {
                float score = es[t] - 2.0f * acc[t][r];
                if (score < minv[r]) { minv[r] = score; mini[r] = n; }
            }
        }

        __syncthreads();   // buffer i&1 free before it is refilled next iter
    }

    // ---- reduce across the 16 lanes holding each row (wave32) ----
    #pragma unroll
    for (int m = 1; m < 16; m <<= 1) {
        #pragma unroll
        for (int r = 0; r < 8; ++r) {
            float ov = __shfl_xor(minv[r], m, 32);
            int   oi = __shfl_xor(mini[r], m, 32);
            if (ov < minv[r] || (ov == minv[r] && oi < mini[r])) {
                minv[r] = ov; mini[r] = oi;
            }
        }
    }
    if (lo == 0) {           // lanes 0 and 16 own rows r and r+8 respectively
        #pragma unroll
        for (int r = 0; r < 8; ++r) {
            int row = rowBase + hi * 8 + r;
            idx_ws[row]     = mini[r];
            idx_out_f[row]  = (float)mini[r];
        }
    }
}

// ---------------------------------------------------------------------------
// Kernel 3: gather z_q (exact f32), commitment loss, segment-sum scatter.
// grid = N_ROWS blocks of 256 threads (one block per row).
// ---------------------------------------------------------------------------
__global__ void vq_scatter(const float* __restrict__ z,
                           const float* __restrict__ E,
                           const int*   __restrict__ idx_ws,
                           float* __restrict__ out_zq,
                           float* __restrict__ out_cs,   // counts accumulator
                           float* __restrict__ out_es,   // embed-sum accumulator
                           float* __restrict__ loss_ws)
{
    const int row = blockIdx.x;
    const int d   = threadIdx.x;
    const int k   = idx_ws[row];

    const float zv = z[(size_t)row * DIM + d];
    const float q  = E[(size_t)k * DIM + d];
    out_zq[(size_t)row * DIM + d] = q;       // z + sg(z_q - z) == z_q

    const float diff = zv - q;
    __shared__ float red[256];
    red[d] = diff * diff;
    __syncthreads();
    #pragma unroll
    for (int s = 128; s > 0; s >>= 1) {
        if (d < s) red[d] += red[d + s];
        __syncthreads();
    }
    if (d == 0) {
        atomicAdd(loss_ws, red[0]);
        atomicAdd(out_cs + k, 1.0f);
    }
    atomicAdd(out_es + (size_t)k * DIM + d, zv);
}

// ---------------------------------------------------------------------------
// Kernel 4: EMA finalize. The scatter sums currently live in the output
// regions out_cs/out_es; blend with old EMA state and overwrite in place.
// grid = K_CODE blocks of 256 threads.
// ---------------------------------------------------------------------------
__global__ void vq_final(const float* __restrict__ old_cs,
                         const float* __restrict__ old_es,
                         const float* __restrict__ loss_ws,
                         float* __restrict__ out_emb,
                         float* __restrict__ out_cs,
                         float* __restrict__ out_es,
                         float* __restrict__ out_loss)
{
    const int k = blockIdx.x;
    const int d = threadIdx.x;
    const int i = k * DIM + d;

    const float cnt = out_cs[k];              // everyone reads counts first
    __syncthreads();                          // ... before lane 0 overwrites

    const float ncs = 0.99f * old_cs[k] + 0.01f * cnt;
    const float nes = 0.99f * old_es[i] + 0.01f * out_es[i];
    out_es[i]  = nes;
    out_emb[i] = nes / (ncs + 1e-5f);
    if (d == 0) out_cs[k] = ncs;
    if (k == 0 && d == 0)
        out_loss[0] = loss_ws[0] * (0.25f / ((float)N_ROWS * (float)DIM));
}

// ---------------------------------------------------------------------------
// Host launcher
// ---------------------------------------------------------------------------
extern "C" void kernel_launch(void* const* d_in, const int* in_sizes, int n_in,
                              void* d_out, int out_size, void* d_ws, size_t ws_size,
                              hipStream_t stream)
{
    const float* z      = (const float*)d_in[0];   // [32768,256]
    const float* E      = (const float*)d_in[1];   // [8192,256]
    const float* ema_cs = (const float*)d_in[2];   // [8192]
    const float* ema_es = (const float*)d_in[3];   // [8192,256]

    float* out = (float*)d_out;
    float* out_zq   = out;                                  // 8388608
    float* out_idx  = out + 8388608;                        // 32768
    float* out_loss = out + 8421376;                        // 1
    float* out_emb  = out + 8421377;                        // 2097152
    float* out_cs   = out + 10518529;                       // 8192
    float* out_es   = out + 10526721;                       // 2097152

    // workspace: Eb(4MB) | e_sq(32KB) | idx(128KB) | loss(4B)  ~= 4.36 MB
    char* ws = (char*)d_ws;
    __bf16* Eb      = (__bf16*)(ws);
    float*  e_sq    = (float*)(ws + 4194304);
    int*    idx_ws  = (int*)  (ws + 4227072);
    float*  loss_ws = (float*)(ws + 4358144);

    vq_prep<<<K_CODE, 256, 0, stream>>>(E, Eb, e_sq, out_cs, out_es, loss_ws);
    vq_argmin<<<N_ROWS / 128, 256, 0, stream>>>(z, Eb, e_sq, idx_ws, out_idx);
    vq_scatter<<<N_ROWS, 256, 0, stream>>>(z, E, idx_ws, out_zq, out_cs, out_es, loss_ws);
    vq_final<<<K_CODE, 256, 0, stream>>>(ema_cs, ema_es, loss_ws,
                                         out_emb, out_cs, out_es, out_loss);
    (void)in_sizes; (void)n_in; (void)out_size; (void)ws_size;
}
